// WaveRNN_86517821211912
// MI455X (gfx1250) — matrix operand order repaired and checked by reading, not compiled, not run on hardware
//
#include <hip/hip_runtime.h>
#include <math.h>

typedef __bf16 bf16;
typedef __attribute__((ext_vector_type(16))) __bf16 v16bf;
typedef __attribute__((ext_vector_type(8)))  float  v8f;
typedef unsigned int v4u __attribute__((ext_vector_type(4)));
typedef int v8i __attribute__((ext_vector_type(8)));
typedef int v4i __attribute__((ext_vector_type(4)));

#define TT   11000
#define BB   8
#define MROW (BB * TT)          // 88000 rows, = 5500 * 16 exactly
#define HH   512
#define GG   1536
#define EPSF 1e-5f
#ifndef USE_TDM
#define USE_TDM 1
#endif

__device__ __forceinline__ bf16 f2bf(float f) {
  unsigned u = __builtin_bit_cast(unsigned, f);
  u += 0x7FFFu + ((u >> 16) & 1u);          // round-to-nearest-even
  unsigned short h = (unsigned short)(u >> 16);
  return __builtin_bit_cast(bf16, h);
}
__device__ __forceinline__ float sigm(float x) { return 1.0f / (1.0f + __expf(-x)); }

// ---------------- conv front-end -------------------------------------------
__global__ void k_conv_in(const float* __restrict__ mels, const float* __restrict__ w,
                          const float* __restrict__ s, const float* __restrict__ bb,
                          const float* __restrict__ m, const float* __restrict__ v,
                          float* __restrict__ out) {
  int idx = blockIdx.x * blockDim.x + threadIdx.x;
  if (idx >= 8 * 128 * 40) return;
  int t = idx % 40, o = (idx / 40) % 128, b = idx / (40 * 128);
  const float* mb = mels + b * 80 * 44;
  const float* wo = w + o * 80 * 5;
  float acc = 0.f;
  for (int c = 0; c < 80; ++c)
    for (int k = 0; k < 5; ++k)
      acc += mb[c * 44 + t + k] * wo[c * 5 + k];
  float val = (acc - m[o]) * (s[o] / sqrtf(v[o] + EPSF)) + bb[o];
  out[idx] = fmaxf(val, 0.f);
}

__global__ void k_pw_bn(const float* __restrict__ in, const float* __restrict__ W,
                        const float* __restrict__ s, const float* __restrict__ bb,
                        const float* __restrict__ m, const float* __restrict__ v,
                        const float* __restrict__ resid, float* __restrict__ out, int relu) {
  int idx = blockIdx.x * blockDim.x + threadIdx.x;
  if (idx >= 8 * 128 * 40) return;
  int t = idx % 40, o = (idx / 40) % 128, b = idx / (40 * 128);
  const float* ib = in + b * 128 * 40 + t;
  const float* wo = W + o * 128;
  float acc = 0.f;
  for (int c = 0; c < 128; ++c) acc += wo[c] * ib[c * 40];
  float val = (acc - m[o]) * (s[o] / sqrtf(v[o] + EPSF)) + bb[o];
  if (relu) val = fmaxf(val, 0.f);
  if (resid) val += resid[idx];
  out[idx] = val;
}

__global__ void k_pw_bias(const float* __restrict__ in, const float* __restrict__ W,
                          const float* __restrict__ bias, float* __restrict__ out) {
  int idx = blockIdx.x * blockDim.x + threadIdx.x;
  if (idx >= 8 * 128 * 40) return;
  int t = idx % 40, o = (idx / 40) % 128, b = idx / (40 * 128);
  const float* ib = in + b * 128 * 40 + t;
  const float* wo = W + o * 128;
  float acc = bias[o];
  for (int c = 0; c < 128; ++c) acc += wo[c] * ib[c * 40];
  out[idx] = acc;
}

__global__ void k_up(const float* __restrict__ in, const float* __restrict__ w,
                     float* __restrict__ out, int Lin, int sc) {
  int Lout = Lin * sc;
  int total = 8 * 80 * Lout;
  int idx = blockIdx.x * blockDim.x + threadIdx.x;
  if (idx >= total) return;
  int o = idx % Lout, f = (idx / Lout) % 80, b = idx / (Lout * 80);
  const float* ib = in + (b * 80 + f) * Lin;
  float acc = 0.f;
  for (int k = 0; k <= 2 * sc; ++k) {
    int i = o + k - sc;
    if (i >= 0 && i < Lout) acc += w[k] * ib[i / sc];
  }
  out[idx] = acc;
}

// ---------------- packing / conversion helpers -----------------------------
__global__ void k_wconv(const float* __restrict__ src, bf16* __restrict__ dst,
                        int N, int K, int Kp) {
  long idx = (long)blockIdx.x * blockDim.x + threadIdx.x;
  long total = (long)N * Kp;
  if (idx >= total) return;
  int k = (int)(idx % Kp); int n = (int)(idx / Kp);
  dst[idx] = (k < K) ? f2bf(src[(long)n * K + k]) : f2bf(0.f);
}

__global__ void k_f2b(const float* __restrict__ src, bf16* __restrict__ dst, long n) {
  long idx = (long)blockIdx.x * blockDim.x + threadIdx.x;
  if (idx < n) dst[idx] = f2bf(src[idx]);
}

__global__ void k_build_xin(const float* __restrict__ x, const float* __restrict__ up2,
                            const float* __restrict__ aux_f, const float* __restrict__ spk,
                            bf16* __restrict__ out) {
  long total = (long)MROW * 384;
  long idx = (long)blockIdx.x * blockDim.x + threadIdx.x;
  if (idx >= total) return;
  int col = (int)(idx % 384); long row = idx / 384;
  int b = (int)(row / TT), t = (int)(row % TT);
  float vv;
  if (col == 0)        vv = x[b * TT + t];
  else if (col < 81)   vv = up2[((long)(b * 80 + (col - 1))) * 12100 + 550 + t];
  else if (col < 113)  vv = aux_f[(b * 128 + (col - 81)) * 40 + t / 275];
  else if (col < 369)  vv = spk[b * 256 + (col - 113)];
  else                 vv = 0.f;
  out[idx] = f2bf(vv);
}

__global__ void k_pack544(float* __restrict__ xr, const float* __restrict__ hist,
                          const float* __restrict__ aux_f, bf16* __restrict__ out,
                          int auxBase, int addHist) {
  long total = (long)MROW * 544;
  long idx = (long)blockIdx.x * blockDim.x + threadIdx.x;
  if (idx >= total) return;
  int col = (int)(idx % 544); long row = idx / 544;
  if (col < 512) {
    float vv = xr[row * 512 + col];
    if (addHist) { vv += hist[row * 512 + col]; xr[row * 512 + col] = vv; }
    out[idx] = f2bf(vv);
  } else {
    int b = (int)(row / TT), t = (int)(row % TT);
    out[idx] = f2bf(aux_f[(b * 128 + auxBase + (col - 512)) * 40 + t / 275]);
  }
}

// ---------------- generic WMMA GEMM ----------------------------------------
// C[M,N] = act(A[M,K]_bf16 @ W[N,K]_bf16^T + bias)
// block = 8 waves; each wave computes 16(M) x 64(N) (4 tiles, A-frag reused 4x);
// grid = (M/16, N/512).
__global__ void __launch_bounds__(256)
k_gemm(const bf16* __restrict__ A, const bf16* __restrict__ W,
       const float* __restrict__ bias, float* __restrict__ C, int N, int K, int relu) {
  int wave = threadIdx.x >> 5;
  int lane = threadIdx.x & 31;
  int grp = lane >> 4, lidx = lane & 15;
  long m0 = (long)blockIdx.x * 16;
  int  n0 = blockIdx.y * 512 + wave * 64;

  const bf16* Ar = A + (m0 + lidx) * K + 16 * grp;
  const bf16* Br[4];
#pragma unroll
  for (int i = 0; i < 4; ++i)
    Br[i] = W + (long)(n0 + i * 16 + lidx) * K + 16 * grp;

  v8f acc[4] = {};
  for (int kt = 0; kt < K; kt += 32) {
    v16bf a;
#pragma unroll
    for (int j = 0; j < 16; ++j) a[j] = Ar[kt + j];
    if (kt + 32 < K) {
      __builtin_prefetch(Ar + kt + 32, 0, 3);
      __builtin_prefetch(Br[0] + kt + 32, 0, 3);
    }
#pragma unroll
    for (int i = 0; i < 4; ++i) {
      v16bf b;
#pragma unroll
      for (int j = 0; j < 16; ++j) b[j] = Br[i][kt + j];
      acc[i] = __builtin_amdgcn_wmma_f32_16x16x32_bf16(false, a, false, b, (short)0,
                                                       acc[i], false, false);
    }
  }
#pragma unroll
  for (int i = 0; i < 4; ++i) {
    int n = n0 + i * 16 + lidx;
    float bv = bias ? bias[n] : 0.f;
    float* Cr = C + (m0 + 8 * grp) * N + n;
#pragma unroll
    for (int r = 0; r < 8; ++r) {
      float vv = acc[i][r] + bv;
      if (relu) vv = fmaxf(vv, 0.f);
      Cr[(long)r * N] = vv;
    }
  }
}

// ---------------- persistent GRU -------------------------------------------
__device__ __forceinline__ void gridbar(unsigned* ctr, unsigned expected) {
  __threadfence();
  __syncthreads();
  if (threadIdx.x == 0) {
    atomicAdd(ctr, 1u);
    while (atomicAdd(ctr, 0u) < expected) __builtin_amdgcn_s_sleep(8);
  }
  __syncthreads();
  __threadfence();
}

// 32 WGs x 96 threads. WG g owns columns [g*16, g*16+16). Wave q in {0,1,2} = gate.
// Wh slice TDM-loaded into LDS once; shared h double-buffered in global as bf16;
// the f32 hidden state for each (b,col) lives in a register of its gate thread.
__global__ void __launch_bounds__(96)
k_gru(const bf16* __restrict__ Wh, const float* __restrict__ bh,
      const float* __restrict__ xW, float* __restrict__ hist,
      bf16* __restrict__ hbuf, unsigned* __restrict__ bar) {
  __shared__ bf16  Ws[48][HH];        // 48KB: rows {q*H + c0 + i}
  __shared__ float ghs[3][16][16];    // gate pre-activations exchange

  int g   = blockIdx.x;
  int c0  = g * 16;
  int tid = threadIdx.x;
  int wave = tid >> 5, lane = tid & 31;
  int grp = lane >> 4, lidx = lane & 15;
  const int NW = gridDim.x;

#if USE_TDM
  {
    // Tensor Data Mover: 2D tile, 16 rows x 512 bf16, row stride 512 elements.
    unsigned long long ga =
        (unsigned long long)(size_t)(Wh + (size_t)(wave * HH + c0) * HH);
    unsigned lds_base = (unsigned)(size_t)(&Ws[wave * 16][0]);
    v4u g0;
    g0[0] = 1u;                                   // count=1, user-mode descriptor
    g0[1] = lds_base;                             // lds_addr
    g0[2] = (unsigned)(ga & 0xFFFFFFFFu);         // global_addr[31:0]
    g0[3] = (unsigned)((ga >> 32) & 0x01FFFFFFu) | (2u << 30);  // addr[56:32], type=2
    v8i g1;
    g1[0] = (int)(1u << 16);                      // data_size=1 (2 bytes)
    g1[1] = (int)(512u << 16);                    // tensor_dim0 = 512 (low 16)
    g1[2] = (int)(16u << 16);                     // dim0 hi=0 | tensor_dim1 = 16
    g1[3] = (int)(512u << 16);                    // dim1 hi=0 | tile_dim0 = 512
    g1[4] = 16;                                   // tile_dim1 = 16, tile_dim2 = 0
    g1[5] = 512;                                  // tensor_dim0_stride[31:0] = 512
    g1[6] = 0;
    g1[7] = 0;
    v4i gz; gz[0] = 0; gz[1] = 0; gz[2] = 0; gz[3] = 0;
#if defined(__clang_major__) && (__clang_major__ >= 23)
    v8i gz8;
    gz8[0] = 0; gz8[1] = 0; gz8[2] = 0; gz8[3] = 0;
    gz8[4] = 0; gz8[5] = 0; gz8[6] = 0; gz8[7] = 0;
    __builtin_amdgcn_tensor_load_to_lds(g0, g1, gz, gz, gz8, 0);
#else
    __builtin_amdgcn_tensor_load_to_lds(g0, g1, gz, gz, 0);
#endif
    __builtin_amdgcn_s_wait_tensorcnt(0);
  }
#else
  for (int idx = tid; idx < 48 * HH; idx += 96) {
    int rl = idx >> 9, k = idx & (HH - 1);
    int q = rl >> 4, i = rl & 15;
    Ws[rl][k] = Wh[(long)(q * HH + c0 + i) * HH + k];
  }
#endif
  __syncthreads();

  float hp0 = 0.f, hp1 = 0.f;   // register-resident f32 hidden state

  for (int t = 0; t < TT; ++t) {
    const bf16* hr = hbuf + (t & 1) * (16 * HH);
    bf16*       hw = hbuf + ((t + 1) & 1) * (16 * HH);

    v8f acc = {};
    const bf16* Wq = &Ws[wave * 16 + lidx][0] + 16 * grp;   // B frag base (LDS)
    const bf16* Ab = hr + lidx * HH + 16 * grp;             // A frag base (global bf16)
    for (int kt = 0; kt < HH; kt += 32) {
      v16bf a, b;
#pragma unroll
      for (int j = 0; j < 16; ++j) a[j] = Ab[kt + j];
#pragma unroll
      for (int j = 0; j < 16; ++j) b[j] = Wq[kt + j];
      acc = __builtin_amdgcn_wmma_f32_16x16x32_bf16(false, a, false, b, (short)0, acc,
                                                    false, false);
    }
    float bv = bh[wave * HH + c0 + lidx];
#pragma unroll
    for (int i = 0; i < 8; ++i) ghs[wave][i + 8 * grp][lidx] = acc[i] + bv;
    __syncthreads();

    // gate fusion: element e = b*16 + j; thread owns e=tid and (tid<32) e=tid+96
    auto gate = [&](int e, float& hp) {
      int b = e >> 4, j = e & 15;
      int col = c0 + j;
      long xrow = ((long)b * TT + t) * GG;
      float r = sigm (xW[xrow +          col] + ghs[0][b][j]);
      float z = sigm (xW[xrow + HH +     col] + ghs[1][b][j]);
      float n = tanhf(xW[xrow + 2 * HH + col] + r * ghs[2][b][j]);
      float hn = (1.f - z) * n + z * hp;
      hp = hn;
      hw[b * HH + col] = f2bf(hn);
      hist[((long)b * TT + t) * HH + col] = hn;
    };
    gate(tid, hp0);
    if (tid < 32) gate(tid + 96, hp1);

    gridbar(bar, (unsigned)((t + 1) * NW));
  }
}

// ---------------- host orchestration ---------------------------------------
static inline int blocks_for(long n, int bs) { return (int)((n + bs - 1) / bs); }

extern "C" void kernel_launch(void* const* d_in, const int* in_sizes, int n_in,
                              void* d_out, int out_size, void* d_ws, size_t ws_size,
                              hipStream_t stream) {
  (void)in_sizes; (void)n_in; (void)out_size; (void)ws_size;
  const float* x        = (const float*)d_in[0];
  const float* mels     = (const float*)d_in[1];
  const float* spk      = (const float*)d_in[2];
  const float* conv_in_w= (const float*)d_in[3];
  const float* bn_in_s  = (const float*)d_in[4];
  const float* bn_in_b  = (const float*)d_in[5];
  const float* bn_in_m  = (const float*)d_in[6];
  const float* bn_in_v  = (const float*)d_in[7];
  const float* res_c1_w = (const float*)d_in[8];
  const float* res_bn1_s= (const float*)d_in[9];
  const float* res_bn1_b= (const float*)d_in[10];
  const float* res_bn1_m= (const float*)d_in[11];
  const float* res_bn1_v= (const float*)d_in[12];
  const float* res_c2_w = (const float*)d_in[13];
  const float* res_bn2_s= (const float*)d_in[14];
  const float* res_bn2_b= (const float*)d_in[15];
  const float* res_bn2_m= (const float*)d_in[16];
  const float* res_bn2_v= (const float*)d_in[17];
  const float* conv_out_w=(const float*)d_in[18];
  const float* conv_out_b=(const float*)d_in[19];
  const float* up_w0    = (const float*)d_in[20];
  const float* up_w1    = (const float*)d_in[21];
  const float* up_w2    = (const float*)d_in[22];
  const float* I_w      = (const float*)d_in[23];
  const float* I_b      = (const float*)d_in[24];
  const float* g1_wi    = (const float*)d_in[25];
  const float* g1_wh    = (const float*)d_in[26];
  const float* g1_bi    = (const float*)d_in[27];
  const float* g1_bh    = (const float*)d_in[28];
  const float* g2_wi    = (const float*)d_in[29];
  const float* g2_wh    = (const float*)d_in[30];
  const float* g2_bi    = (const float*)d_in[31];
  const float* g2_bh    = (const float*)d_in[32];
  const float* fc1_w    = (const float*)d_in[33];
  const float* fc1_b    = (const float*)d_in[34];
  const float* fc2_w    = (const float*)d_in[35];
  const float* fc2_b    = (const float*)d_in[36];
  const float* fc3_w    = (const float*)d_in[37];
  const float* fc3_b    = (const float*)d_in[38];
  float* out = (float*)d_out;

  // ---- workspace carve ----
  char* p = (char*)d_ws;
  auto alloc = [&](size_t bytes) -> char* {
    char* r = p; p += (bytes + 255) & ~(size_t)255; return r;
  };
  float* h_conv = (float*)alloc(8 * 128 * 40 * 4);
  float* t_conv = (float*)alloc(8 * 128 * 40 * 4);
  float* aux_f  = (float*)alloc(8 * 128 * 40 * 4);
  float* up0    = (float*)alloc(8 * 80 * 220  * 4);
  float* up1    = (float*)alloc(8 * 80 * 1100 * 4);
  float* up2    = (float*)alloc((size_t)8 * 80 * 12100 * 4);
  bf16* Iw_bf   = (bf16*)alloc((size_t)512 * 384 * 2);
  bf16* g1wi_bf = (bf16*)alloc((size_t)1536 * 512 * 2);
  bf16* g1wh_bf = (bf16*)alloc((size_t)1536 * 512 * 2);
  bf16* g2wi_bf = (bf16*)alloc((size_t)1536 * 544 * 2);
  bf16* g2wh_bf = (bf16*)alloc((size_t)1536 * 512 * 2);
  bf16* fc1_bf  = (bf16*)alloc((size_t)512 * 544 * 2);
  bf16* fc2_bf  = (bf16*)alloc((size_t)512 * 544 * 2);
  bf16* fc3_bf  = (bf16*)alloc((size_t)512 * 512 * 2);
  bf16* xin_bf  = (bf16*)alloc((size_t)MROW * 384 * 2);
  float* xi_f   = (float*)alloc((size_t)MROW * 512 * 4);
  bf16* xi_bf   = (bf16*)alloc((size_t)MROW * 512 * 2);
  float* xW     = (float*)alloc((size_t)MROW * 1536 * 4);
  float* hist   = (float*)alloc((size_t)MROW * 512 * 4);
  bf16* in2_bf  = (bf16*)alloc((size_t)MROW * 544 * 2);
  float* f_f    = (float*)alloc((size_t)MROW * 512 * 4);
  bf16* hbuf    = (bf16*)alloc(2 * 16 * 512 * 2);
  unsigned* bar = (unsigned*)alloc(256);

  const int BS = 256;

  // 1) conv front-end
  k_conv_in<<<blocks_for(8 * 128 * 40, BS), BS, 0, stream>>>(
      mels, conv_in_w, bn_in_s, bn_in_b, bn_in_m, bn_in_v, h_conv);
  for (int i = 0; i < 10; ++i) {
    k_pw_bn<<<blocks_for(8 * 128 * 40, BS), BS, 0, stream>>>(
        h_conv, res_c1_w + (size_t)i * 128 * 128,
        res_bn1_s + i * 128, res_bn1_b + i * 128, res_bn1_m + i * 128, res_bn1_v + i * 128,
        nullptr, t_conv, 1);
    k_pw_bn<<<blocks_for(8 * 128 * 40, BS), BS, 0, stream>>>(
        t_conv, res_c2_w + (size_t)i * 128 * 128,
        res_bn2_s + i * 128, res_bn2_b + i * 128, res_bn2_m + i * 128, res_bn2_v + i * 128,
        h_conv, h_conv, 0);
  }
  k_pw_bias<<<blocks_for(8 * 128 * 40, BS), BS, 0, stream>>>(h_conv, conv_out_w, conv_out_b, aux_f);

  // 2) mel upsampling 44 -> 220 -> 1100 -> 12100
  k_up<<<blocks_for((long)8 * 80 * 220, BS), BS, 0, stream>>>(mels, up_w0, up0, 44, 5);
  k_up<<<blocks_for((long)8 * 80 * 1100, BS), BS, 0, stream>>>(up0, up_w1, up1, 220, 5);
  k_up<<<blocks_for((long)8 * 80 * 12100, BS), BS, 0, stream>>>(up1, up_w2, up2, 1100, 11);

  // 3) weight conversions (f32 -> bf16, K zero-padded where needed)
  k_wconv<<<blocks_for((long)512 * 384, BS), BS, 0, stream>>>(I_w, Iw_bf, 512, 369, 384);
  k_wconv<<<blocks_for((long)1536 * 512, BS), BS, 0, stream>>>(g1_wi, g1wi_bf, 1536, 512, 512);
  k_wconv<<<blocks_for((long)1536 * 512, BS), BS, 0, stream>>>(g1_wh, g1wh_bf, 1536, 512, 512);
  k_wconv<<<blocks_for((long)1536 * 544, BS), BS, 0, stream>>>(g2_wi, g2wi_bf, 1536, 544, 544);
  k_wconv<<<blocks_for((long)1536 * 512, BS), BS, 0, stream>>>(g2_wh, g2wh_bf, 1536, 512, 512);
  k_wconv<<<blocks_for((long)512 * 544, BS), BS, 0, stream>>>(fc1_w, fc1_bf, 512, 544, 544);
  k_wconv<<<blocks_for((long)512 * 544, BS), BS, 0, stream>>>(fc2_w, fc2_bf, 512, 544, 544);
  k_wconv<<<blocks_for((long)512 * 512, BS), BS, 0, stream>>>(fc3_w, fc3_bf, 512, 512, 512);

  // 4) xi = I(xin)
  k_build_xin<<<blocks_for((long)MROW * 384, BS), BS, 0, stream>>>(x, up2, aux_f, spk, xin_bf);
  k_gemm<<<dim3(MROW / 16, 1), 256, 0, stream>>>(xin_bf, Iw_bf, I_b, xi_f, 512, 384, 0);
  k_f2b<<<blocks_for((long)MROW * 512, BS), BS, 0, stream>>>(xi_f, xi_bf, (long)MROW * 512);

  // 5) GRU1
  k_gemm<<<dim3(MROW / 16, 3), 256, 0, stream>>>(xi_bf, g1wi_bf, g1_bi, xW, 1536, 512, 0);
  hipMemsetAsync(hbuf, 0, 2 * 16 * 512 * sizeof(bf16), stream);
  hipMemsetAsync(bar, 0, sizeof(unsigned), stream);
  k_gru<<<dim3(32), dim3(96), 0, stream>>>(g1wh_bf, g1_bh, xW, hist, hbuf, bar);

  // 6) xr = xi + h1 ; in2 = [xr | a2] ; GRU2
  k_pack544<<<blocks_for((long)MROW * 544, BS), BS, 0, stream>>>(xi_f, hist, aux_f, in2_bf, 32, 1);
  k_gemm<<<dim3(MROW / 16, 3), 256, 0, stream>>>(in2_bf, g2wi_bf, g2_bi, xW, 1536, 544, 0);
  hipMemsetAsync(hbuf, 0, 2 * 16 * 512 * sizeof(bf16), stream);
  hipMemsetAsync(bar, 0, sizeof(unsigned), stream);
  k_gru<<<dim3(32), dim3(96), 0, stream>>>(g2wh_bf, g2_bh, xW, hist, hbuf, bar);

  // 7) FC stack
  k_pack544<<<blocks_for((long)MROW * 544, BS), BS, 0, stream>>>(xi_f, hist, aux_f, in2_bf, 64, 1);
  k_gemm<<<dim3(MROW / 16, 1), 256, 0, stream>>>(in2_bf, fc1_bf, fc1_b, f_f, 512, 544, 1);
  k_pack544<<<blocks_for((long)MROW * 544, BS), BS, 0, stream>>>(f_f, nullptr, aux_f, in2_bf, 96, 0);
  k_gemm<<<dim3(MROW / 16, 1), 256, 0, stream>>>(in2_bf, fc2_bf, fc2_b, f_f, 512, 544, 1);
  k_f2b<<<blocks_for((long)MROW * 512, BS), BS, 0, stream>>>(f_f, xi_bf, (long)MROW * 512);
  k_gemm<<<dim3(MROW / 16, 1), 256, 0, stream>>>(xi_bf, fc3_bf, fc3_b, out, 512, 512, 0);
}
